// GraphFingerprint_59390807769385
// MI455X (gfx1250) — compile-verified, matrix operand order checked by reading, not run on hardware
//
#include <hip/hip_runtime.h>
#include <hip/hip_bf16.h>
#include <math.h>

// ---------------------------------------------------------------------------
// CDNA5 WMMA (wave32): v_wmma_f32_16x16x32_bf16
// Lane l (half = l>>4, m = l&15) holds A-row m elements
//   k = half*8..+7  and  k = 16+half*8..+7  (two contiguous 16B runs)
// -> stage bf16 tiles row-major in LDS, fragments = two ds_load_b128.
// B is staged from a pre-transposed bf16 weight matrix Wt[N,K], so both
// operands are contiguous 16B-per-lane copies -> async global->LDS DMA.
// ---------------------------------------------------------------------------
typedef __attribute__((ext_vector_type(16))) __bf16 bf16x16;
typedef __attribute__((ext_vector_type(2)))  __bf16 bf16x2;
typedef __attribute__((ext_vector_type(8)))  float  f32x8;
typedef unsigned short ushort_t;

// builtin signature (from hipcc diagnostic): params are v4i32* in AS(1)/AS(3)
typedef int v4i_gcc __attribute__((vector_size(16)));
typedef __attribute__((address_space(1))) v4i_gcc* as1_v4i_ptr;
typedef __attribute__((address_space(3))) v4i_gcc* as3_v4i_ptr;

#if defined(__HIP_DEVICE_COMPILE__) && \
    __has_builtin(__builtin_amdgcn_global_load_async_to_lds_b128) && \
    __has_builtin(__builtin_amdgcn_s_wait_asynccnt)
#define USE_ASYNC_LDS 1
#define WAIT_ASYNC(n) __builtin_amdgcn_s_wait_asynccnt(n)
#else
#define USE_ASYNC_LDS 0
#define WAIT_ASYNC(n)
#endif

static __device__ __forceinline__ void async_cp16(void* lds, const void* g) {
#if USE_ASYNC_LDS
    __builtin_amdgcn_global_load_async_to_lds_b128(
        (as1_v4i_ptr)(g), (as3_v4i_ptr)(lds), 0, 0);
#else
    *(uint4*)lds = *(const uint4*)g;
#endif
}

static __device__ __forceinline__ unsigned int pk_bf16(float a, float b) {
    bf16x2 p = { (__bf16)a, (__bf16)b };          // lowers to v_cvt_pk_bf16_f32
    return __builtin_bit_cast(unsigned int, p);
}

// ---------------------------------------------------------------------------
// Small utility kernels
// ---------------------------------------------------------------------------
__global__ void k_fill(float* __restrict__ p, float v, long n) {
    long i = (long)blockIdx.x * blockDim.x + threadIdx.x;
    if (i < n) p[i] = v;
}

__global__ void k_degcount(const int* __restrict__ dst, float* __restrict__ deg, int E) {
    int e = blockIdx.x * blockDim.x + threadIdx.x;
    if (e < E) atomicAdd(&deg[dst[e]], 1.0f);
}

__global__ void k_rsqrt(float* __restrict__ d, int n) {
    int i = blockIdx.x * blockDim.x + threadIdx.x;
    if (i < n) d[i] = rsqrtf(d[i]);
}

// fp32 -> packed bf16 (n even)
__global__ void k_f32_to_bf16(const float* __restrict__ in, ushort_t* __restrict__ out, long n) {
    long i = ((long)blockIdx.x * blockDim.x + threadIdx.x) * 2;
    if (i < n)
        *(unsigned int*)(out + i) = pk_bf16(in[i], in[i + 1]);
}

// W [K,Nc] f32 -> Wt [Nc,K] bf16 (transpose + convert; weights are small)
__global__ void k_transpose_bf16(const float* __restrict__ W, ushort_t* __restrict__ Wt,
                                 int K, int Nc) {
    long idx = (long)blockIdx.x * blockDim.x + threadIdx.x;
    if (idx >= (long)K * Nc) return;
    int k = (int)(idx / Nc), n = (int)(idx % Nc);
    __bf16 v = (__bf16)W[idx];
    Wt[(long)n * K + k] = __builtin_bit_cast(ushort_t, v);
}

// out[i,f] = h[i,f] * dis[i]^2   (self-loop contribution, initializes out)
__global__ void k_agg_self(const float* __restrict__ h, const float* __restrict__ dis,
                           float* __restrict__ out, int Nn, int F) {
    long idx = (long)blockIdx.x * blockDim.x + threadIdx.x;
    if (idx >= (long)Nn * F) return;
    int i = (int)(idx / F);
    float s = dis[i];
    out[idx] = h[idx] * s * s;
}

// out[dst,f] += h[src,f] * dis[src]*dis[dst]   (one block per edge)
__global__ void k_agg_edges(const int* __restrict__ src, const int* __restrict__ dst,
                            const float* __restrict__ h, const float* __restrict__ dis,
                            float* __restrict__ out, int F) {
    int e = blockIdx.x;
    int s = src[e], d = dst[e];
    float nrm = dis[s] * dis[d];
    const float* hs = h + (long)s * F;
    float* od = out + (long)d * F;
    for (int f = threadIdx.x; f < F; f += blockDim.x)
        atomicAdd(&od[f], hs[f] * nrm);
}

__global__ void k_bias_relu(float* __restrict__ x, const float* __restrict__ b, int Nn, int F) {
    long idx = (long)blockIdx.x * blockDim.x + threadIdx.x;
    if (idx >= (long)Nn * F) return;
    int f = (int)(idx % F);
    x[idx] = fmaxf(x[idx] + b[f], 0.0f);
}

// ---------------------------------------------------------------------------
// WMMA GEMM (bf16 operands): C[M,Nc] = A[M,K] @ Bt[Nc,K]^T (+ bias, relu)
// Block 256 threads = 8 waves; block tile 64x64; K-step 32; LDS double-buffer.
// Wave w: wm = w&3 row subtile, wn = w>>2 col half; 2 WMMAs share A fragment.
// Staging: 16B contiguous per thread per operand -> async DMA to LDS
// (GLOBAL_LOAD_ASYNC_TO_LDS_B128 + s_wait_asynccnt), sync-copy fallback.
// Requires: K % 32 == 0, Nc % 64 == 0; M guarded via source-row clamp +
// guarded epilogue stores.
// ---------------------------------------------------------------------------
__global__ void __launch_bounds__(256)
k_gemm_wmma_bf16(const ushort_t* __restrict__ A,   // [M,K] bf16 row-major
                 const ushort_t* __restrict__ Bt,  // [Nc,K] bf16 row-major
                 const float* __restrict__ bias, float* __restrict__ C,
                 int M, int K, int Nc, int relu) {
    __shared__ uint4 lA[2][64 * 4];   // [buf][row][4 x 16B]  (32 bf16 per row)
    __shared__ uint4 lB[2][64 * 4];   // [buf][n  ][4 x 16B]  (k-major)

    const int tid  = threadIdx.x;
    const int lane = tid & 31;
    const int wave = tid >> 5;
    const int wm   = wave & 3;
    const int wn   = wave >> 2;
    const int half = lane >> 4;
    const int mrow = lane & 15;

    const int  blockN = blockIdx.x * 64;
    const long blockM = (long)blockIdx.y * 64;

    // staging coords: one 16B chunk of one row per operand per thread
    const int arow = tid >> 2;          // 0..63
    const int a16  = tid & 3;           // chunk (k-group of 8)
    const int bn   = tid & 63;          // 0..63
    const int b16  = tid >> 6;          // chunk

    long garow = blockM + arow;
    if (garow >= M) garow = M - 1;      // clamp: rows >= M discarded at store
    const ushort_t* aRow = A  + garow * (long)K + a16 * 8;
    const ushort_t* bRow = Bt + (long)(blockN + bn) * K + b16 * 8;

    uint4* lAslot0 = &lA[0][arow * 4 + a16];
    uint4* lBslot0 = &lB[0][bn * 4 + b16];
    uint4* lAslot1 = &lA[1][arow * 4 + a16];
    uint4* lBslot1 = &lB[1][bn * 4 + b16];

    const int fragRowA = (wm * 16 + mrow) * 4;
    const int fragRowB0 = (wn * 32 + mrow) * 4;
    const int fragRowB1 = (wn * 32 + 16 + mrow) * 4;

    f32x8 acc0 = {};
    f32x8 acc1 = {};

    const int nt = K >> 5;
    // preload tile 0
    async_cp16(lAslot0, aRow);
    async_cp16(lBslot0, bRow);

    for (int t = 0; t < nt; ++t) {
        const int cur = t & 1;
        if (t + 1 < nt) {               // stage next tile into other buffer
            const int k1 = (t + 1) << 5;
            async_cp16(cur ? lAslot0 : lAslot1, aRow + k1);
            async_cp16(cur ? lBslot0 : lBslot1, bRow + k1);
            WAIT_ASYNC(2);              // previous tile's 2 DMAs done
        } else {
            WAIT_ASYNC(0);
        }
        __syncthreads();                // all waves' DMAs visible

        bf16x16 af, bf0, bf1;
        {
            uint4* p = reinterpret_cast<uint4*>(&af);
            p[0] = lA[cur][fragRowA + half];
            p[1] = lA[cur][fragRowA + 2 + half];
        }
        {
            uint4* p = reinterpret_cast<uint4*>(&bf0);
            p[0] = lB[cur][fragRowB0 + half];
            p[1] = lB[cur][fragRowB0 + 2 + half];
        }
        {
            uint4* p = reinterpret_cast<uint4*>(&bf1);
            p[0] = lB[cur][fragRowB1 + half];
            p[1] = lB[cur][fragRowB1 + 2 + half];
        }
        acc0 = __builtin_amdgcn_wmma_f32_16x16x32_bf16(
                   false, af, false, bf0, (short)0, acc0, false, false);
        acc1 = __builtin_amdgcn_wmma_f32_16x16x32_bf16(
                   false, af, false, bf1, (short)0, acc1, false, false);
        __syncthreads();                // reads done before buffer reuse
    }

    const int col0 = blockN + wn * 32 + mrow;
    const int col1 = col0 + 16;
    const float bv0 = bias ? bias[col0] : 0.0f;
    const float bv1 = bias ? bias[col1] : 0.0f;
    #pragma unroll
    for (int r = 0; r < 8; ++r) {
        long row = blockM + wm * 16 + half * 8 + r;
        if (row < M) {
            float v0 = acc0[r] + bv0;
            float v1 = acc1[r] + bv1;
            if (relu) { v0 = fmaxf(v0, 0.0f); v1 = fmaxf(v1, 0.0f); }
            C[row * (long)Nc + col0] = v0;
            C[row * (long)Nc + col1] = v1;
        }
    }
}

// ---------------------------------------------------------------------------
// gate[i] = sigmoid(dot(x[i,:], pw) + pb)  -- one wave32 per node
// ---------------------------------------------------------------------------
__global__ void k_gate(const float* __restrict__ x, const float* __restrict__ pw,
                       const float* __restrict__ pb, float* __restrict__ gate,
                       int Nn, int F) {
    int wave = threadIdx.x >> 5, lane = threadIdx.x & 31;
    int i = blockIdx.x * 8 + wave;
    if (i >= Nn) return;
    const float* xi = x + (long)i * F;
    float s = 0.0f;
    for (int f = lane; f < F; f += 32) s += xi[f] * pw[f];
    #pragma unroll
    for (int off = 16; off > 0; off >>= 1) s += __shfl_xor(s, off, 32);
    if (lane == 0) gate[i] = 1.0f / (1.0f + __expf(-(s + pb[0])));
}

// ---------------------------------------------------------------------------
// pooled weighted sum + pooled max into feats[g, 0:F] / feats[g, F:2F]
// (x >= 0 post-ReLU, so int-bitcast atomicMax against 0-init is exact)
// ---------------------------------------------------------------------------
__global__ void k_pool(const float* __restrict__ x, const float* __restrict__ gate,
                       const int* __restrict__ batch, float* __restrict__ feats,
                       int Nn, int F) {
    long idx = (long)blockIdx.x * blockDim.x + threadIdx.x;
    if (idx >= (long)Nn * F) return;
    int i = (int)(idx / F);
    int f = (int)(idx % F);
    int g = batch[i];
    float v = x[idx];
    float* row = feats + (long)g * (2 * F);
    atomicAdd(&row[f], v * gate[i]);
    atomicMax((int*)&row[F + f], __float_as_int(v));
}

// ---------------------------------------------------------------------------
extern "C" void kernel_launch(void* const* d_in, const int* in_sizes, int n_in,
                              void* d_out, int out_size, void* d_ws, size_t ws_size,
                              hipStream_t stream) {
    const float* x0 = (const float*)d_in[0];
    const int* eidx = (const int*)d_in[1];
    const int* batch = (const int*)d_in[2];
    const float* W1 = (const float*)d_in[3]; const float* b1 = (const float*)d_in[4];
    const float* W2 = (const float*)d_in[5]; const float* b2 = (const float*)d_in[6];
    const float* W3 = (const float*)d_in[7]; const float* b3 = (const float*)d_in[8];
    const float* pw = (const float*)d_in[9]; const float* pb = (const float*)d_in[10];
    const float* Wo = (const float*)d_in[11]; const float* bo = (const float*)d_in[12];

    const int Nn  = in_sizes[2];
    const int Fin = in_sizes[0] / Nn;
    const int E   = in_sizes[1] / 2;
    const int H0  = in_sizes[4];
    const int H1  = in_sizes[6];
    const int H2  = in_sizes[8];
    const int FP  = in_sizes[12];
    const int G   = out_size / FP;

    const int* src = eidx;
    const int* dst = eidx + E;

    // byte-addressed workspace carve-out, 256B aligned
    char* base = (char*)d_ws;
    size_t woff = 0;
    auto alloc = [&](size_t bytes) {
        woff = (woff + 255) & ~(size_t)255;
        void* p = base + woff;
        woff += bytes;
        return p;
    };
    float*    deg   = (float*)alloc((size_t)Nn * 4);
    float*    gate  = (float*)alloc((size_t)Nn * 4);
    float*    xb    = (float*)alloc((size_t)Nn * H2 * 4);       // activation ping
    float*    hb    = (float*)alloc((size_t)Nn * H2 * 4);       // pre-agg pong
    float*    feats = (float*)alloc((size_t)G * 2 * H2 * 4);
    ushort_t* Abf   = (ushort_t*)alloc((size_t)Nn * H2 * 2);    // bf16 activations
    ushort_t* fbf   = (ushort_t*)alloc((size_t)G * 2 * H2 * 2); // bf16 feats
    ushort_t* Wt1   = (ushort_t*)alloc((size_t)Fin * H0 * 2);
    ushort_t* Wt2   = (ushort_t*)alloc((size_t)H0 * H1 * 2);
    ushort_t* Wt3   = (ushort_t*)alloc((size_t)H1 * H2 * 2);
    ushort_t* Wto   = (ushort_t*)alloc((size_t)(2 * H2) * FP * 2);
    (void)ws_size;

    const int T = 256;
    auto blocks = [](long n, int t) { return (unsigned)((n + t - 1) / t); };

    // ---- pre-transpose + convert all weights to bf16 [N,K] ----
    k_transpose_bf16<<<blocks((long)Fin * H0, T), T, 0, stream>>>(W1, Wt1, Fin, H0);
    k_transpose_bf16<<<blocks((long)H0 * H1, T), T, 0, stream>>>(W2, Wt2, H0, H1);
    k_transpose_bf16<<<blocks((long)H1 * H2, T), T, 0, stream>>>(W3, Wt3, H1, H2);
    k_transpose_bf16<<<blocks((long)(2 * H2) * FP, T), T, 0, stream>>>(Wo, Wto, 2 * H2, FP);

    // ---- degree -> dis = rsqrt(1 + indeg) ----
    k_fill<<<blocks(Nn, T), T, 0, stream>>>(deg, 1.0f, Nn);
    k_degcount<<<blocks(E, T), T, 0, stream>>>(dst, deg, E);
    k_rsqrt<<<blocks(Nn, T), T, 0, stream>>>(deg, Nn);

    auto gemm = [&](const ushort_t* A, const ushort_t* Bt, const float* bias, float* C,
                    int M, int K, int Ncols, int relu) {
        dim3 grid(Ncols / 64, (M + 63) / 64);
        k_gemm_wmma_bf16<<<grid, dim3(256), 0, stream>>>(A, Bt, bias, C, M, K, Ncols, relu);
    };
    auto layer = [&](const float* xin, const ushort_t* Wt, const float* b,
                     int Kin, int Fout) {
        k_f32_to_bf16<<<blocks((long)Nn * Kin / 2, T), T, 0, stream>>>(xin, Abf, (long)Nn * Kin);
        gemm(Abf, Wt, nullptr, hb, Nn, Kin, Fout, 0);
        long nf = (long)Nn * Fout;
        k_agg_self<<<blocks(nf, T), T, 0, stream>>>(hb, deg, xb, Nn, Fout);
        k_agg_edges<<<E, 128, 0, stream>>>(src, dst, hb, deg, xb, Fout);
        k_bias_relu<<<blocks(nf, T), T, 0, stream>>>(xb, b, Nn, Fout);
    };

    layer(x0, Wt1, b1, Fin, H0);
    layer(xb, Wt2, b2, H0, H1);
    layer(xb, Wt3, b3, H1, H2);

    // ---- pooling ----
    k_gate<<<blocks((long)Nn, 8), 256, 0, stream>>>(xb, pw, pb, gate, Nn, H2);
    k_fill<<<blocks((long)G * 2 * H2, T), T, 0, stream>>>(feats, 0.0f, (long)G * 2 * H2);
    k_pool<<<blocks((long)Nn * H2, T), T, 0, stream>>>(xb, gate, batch, feats, Nn, H2);

    // ---- head: relu(feats @ Wo + bo) -> d_out [G, FP] ----
    k_f32_to_bf16<<<blocks((long)G * 2 * H2 / 2, T), T, 0, stream>>>(feats, fbf, (long)G * 2 * H2);
    gemm(fbf, Wto, bo, (float*)d_out, G, 2 * H2, FP, 1);
}
